// MeshGraphNet_7335804142388
// MI455X (gfx1250) — matrix operand order repaired and checked by reading, not compiled, not run on hardware
//
#include <hip/hip_runtime.h>

// MeshGraphNet on gfx1250: all linear layers on V_WMMA_F32_16X16X32_F16.
// 256-row tiles per workgroup, 8 wave32s x 32 rows each (two A frags share
// every B frag -> half the LDS B-traffic per MAC), 4-layer MLP fused in LDS.

#define NNODES 50000
#define NEDGES 800000
#define BROWS  256
#define TPB    256
#define SLOPE  0.01f
#define LNEPS  1e-5f

typedef _Float16 half_t;
typedef __attribute__((ext_vector_type(16))) _Float16 v16h;
typedef __attribute__((ext_vector_type(8)))  float    v8f;

struct MlpP {
  const float *Wi, *bi, *Wh0, *bh0, *Wh1, *bh1, *Wo, *bo, *g, *b;
};

// ---- dynamic LDS layout (208 KB total) ----
// [0,48K)      : transposed weights (half)  -- max layer = 128 x 192
// [48K,144K)   : act buffer A0 (half)       -- max 256 x 192 ; aliased fp32 final (256x64)
// [144K,208K)  : act buffer A1 (half)       -- 256 x 128
#define OFF_W   0
#define OFF_A0  49152
#define OFF_A1  147456
#define SMEM_BYTES 212992

// A fragment (16x32 f16): lane m=l&15 holds row m; VGPR half-split:
// elems 0..7 -> K = kk + hf*8 + j ; elems 8..15 -> K = kk + 16 + hf*8 + (j&7)
__device__ inline v16h load_a_frag(const half_t* act, int lda, int row16, int kk, int lane) {
  int m = lane & 15, hf = lane >> 4;
  const half_t* p = act + (row16 + m) * lda + kk + hf * 8;
  union { uint4 q[2]; v16h v; } u;
  u.q[0] = *(const uint4*)(p);
  u.q[1] = *(const uint4*)(p + 16);
  return u.v;
}

// B fragment (32x16 f16) from transposed weights wT[N][ldw]:
// lane n=l&15 holds column n; elems 0..7 -> K = kk + hf*8 + j ; elems 8..15 -> K = kk + 16 + hf*8 + (j&7)
__device__ inline v16h load_b_frag(const half_t* wT, int ldw, int col16, int kk, int lane) {
  int n = lane & 15, hf = lane >> 4;
  const half_t* p = wT + (col16 + n) * ldw + kk + hf * 8;
  union { uint4 q[2]; v16h v; } u;
  u.q[0] = *(const uint4*)(p);
  u.q[1] = *(const uint4*)(p + 16);
  return u.v;
}

// Each wave owns 32 rows: two A fragments per k-step share each B fragment.
template <int NT>
__device__ inline void wave_gemm(const half_t* act, int lda, const half_t* wT, int ldw,
                                 int KP, v8f accLo[NT], v8f accHi[NT]) {
  const int lane  = threadIdx.x & 31;
  const int row32 = (threadIdx.x >> 5) * 32;   // 8 waves x 32 rows = 256 rows
  for (int kk = 0; kk < KP; kk += 32) {
    v16h a0 = load_a_frag(act, lda, row32,      kk, lane);
    v16h a1 = load_a_frag(act, lda, row32 + 16, kk, lane);
#pragma unroll
    for (int t = 0; t < NT; ++t) {
      v16h b = load_b_frag(wT, ldw, t * 16, kk, lane);
      accLo[t] = __builtin_amdgcn_wmma_f32_16x16x32_f16(
          false, a0, false, b, (short)0, accLo[t], false, false);
      accHi[t] = __builtin_amdgcn_wmma_f32_16x16x32_f16(
          false, a1, false, b, (short)0, accHi[t], false, false);
    }
  }
}

// Stage global fp32 W[Krows][Ncols] -> LDS half wT[Npad][KP], transposed + zero padded.
__device__ inline void stage_wT(const float* W, int Krows, int Ncols, int Npad, int KP,
                                half_t* wT) {
  int total = Npad * KP;
  for (int idx = threadIdx.x; idx < total; idx += TPB) {
    int n = idx / KP, k = idx - n * KP;
    float v = (k < Krows && n < Ncols) ? W[k * Ncols + n] : 0.0f;
    wT[idx] = (half_t)v;
  }
}

template <int NT>
__device__ inline void epilogue_leaky(v8f accLo[NT], v8f accHi[NT], const float* bias,
                                      half_t* out, int ldo) {
  const int lane = threadIdx.x & 31;
  const int r0 = (threadIdx.x >> 5) * 32 + ((lane >> 4) << 3);  // C: vgpr r -> row r0+r
  const int c0 = lane & 15;
#pragma unroll
  for (int t = 0; t < NT; ++t) {
    float bv = bias[t * 16 + c0];
#pragma unroll
    for (int r = 0; r < 8; ++r) {
      float v = accLo[t][r] + bv;
      v = v > 0.f ? v : SLOPE * v;
      out[(r0 + r) * ldo + t * 16 + c0] = (half_t)v;
      float w = accHi[t][r] + bv;
      w = w > 0.f ? w : SLOPE * w;
      out[(r0 + 16 + r) * ldo + t * 16 + c0] = (half_t)w;
    }
  }
}

template <int NT>
__device__ inline void epilogue_f32(v8f accLo[NT], v8f accHi[NT], const float* bias,
                                    int Ncols, float* out, int ldo) {
  const int lane = threadIdx.x & 31;
  const int r0 = (threadIdx.x >> 5) * 32 + ((lane >> 4) << 3);
  const int c0 = lane & 15;
#pragma unroll
  for (int t = 0; t < NT; ++t) {
    int c = t * 16 + c0;
    float bv = (c < Ncols) ? bias[c] : 0.f;
#pragma unroll
    for (int r = 0; r < 8; ++r) {
      out[(r0 + r) * ldo + c]      = accLo[t][r] + bv;
      out[(r0 + 16 + r) * ldo + c] = accHi[t][r] + bv;
    }
  }
}

// 4-layer MLP: A0(KP1) ->L1-> A1(128) ->L2-> A0(128) ->L3-> A1(128) ->L4-> fp32 FO (A0 alias)
template <int KP1, int NTOUT>
__device__ inline void mlp_core(const MlpP P, int Kin, int NoutCols, char* smem) {
  half_t* W  = (half_t*)(smem + OFF_W);
  half_t* A0 = (half_t*)(smem + OFF_A0);
  half_t* A1 = (half_t*)(smem + OFF_A1);
  float*  FO = (float*)(smem + OFF_A0);

  stage_wT(P.Wi, Kin, 128, 128, KP1, W);
  __syncthreads();  // also covers caller's input staging into A0
  { v8f aL[8] = {}, aH[8] = {};
    wave_gemm<8>(A0, KP1, W, KP1, KP1, aL, aH);
    epilogue_leaky<8>(aL, aH, P.bi, A1, 128); }
  __syncthreads();

  stage_wT(P.Wh0, 128, 128, 128, 128, W);
  __syncthreads();
  { v8f aL[8] = {}, aH[8] = {};
    wave_gemm<8>(A1, 128, W, 128, 128, aL, aH);
    epilogue_leaky<8>(aL, aH, P.bh0, A0, 128); }
  __syncthreads();

  stage_wT(P.Wh1, 128, 128, 128, 128, W);
  __syncthreads();
  { v8f aL[8] = {}, aH[8] = {};
    wave_gemm<8>(A0, 128, W, 128, 128, aL, aH);
    epilogue_leaky<8>(aL, aH, P.bh1, A1, 128); }
  __syncthreads();

  stage_wT(P.Wo, 128, NoutCols, NTOUT * 16, 128, W);
  __syncthreads();
  { v8f aL[NTOUT] = {}, aH[NTOUT] = {};
    wave_gemm<NTOUT>(A1, 128, W, 128, 128, aL, aH);
    epilogue_f32<NTOUT>(aL, aH, P.bo, NoutCols, FO, NTOUT * 16); }
  __syncthreads();
}

__device__ inline void ln_stats(const float* f, float& mu, float& inv) {
  mu = 0.f;
  for (int c = 0; c < 64; ++c) mu += f[c];
  mu *= (1.f / 64.f);
  float var = 0.f;
  for (int c = 0; c < 64; ++c) { float d = f[c] - mu; var += d * d; }
  inv = rsqrtf(var * (1.f / 64.f) + LNEPS);
}

// ---------------- kernels ----------------

__global__ __launch_bounds__(TPB) void enc_node_kernel(const float* nfeat, const float* flow,
                                                       const int* inlet, MlpP P, half_t* h,
                                                       int nRows) {
  extern __shared__ char smem[];
  half_t* A0 = (half_t*)(smem + OFF_A0);
  int base = blockIdx.x * BROWS;
  for (int idx = threadIdx.x; idx < BROWS * 32; idx += TPB) {  // KP=32 (pad 13)
    int r = idx >> 5, k = idx & 31;
    int row = base + r;
    float v = 0.f;
    if (row < nRows) {
      if (k < 12) v = nfeat[row * 12 + k];
      else if (k == 12) v = inlet[row] ? flow[row] : 0.f;
    }
    A0[idx] = (half_t)v;
  }
  mlp_core<32, 4>(P, 13, 64, smem);
  float* FO = (float*)(smem + OFF_A0);
  int r = threadIdx.x;
  int row = base + r;
  if (row < nRows) {
    const float* f = FO + r * 64;
    float mu, inv; ln_stats(f, mu, inv);
    for (int c = 0; c < 64; ++c)
      h[row * 64 + c] = (half_t)((f[c] - mu) * inv * P.g[c] + P.b[c]);
  }
}

__global__ __launch_bounds__(TPB) void enc_edge_kernel(const float* efeat, MlpP P, half_t* e,
                                                       int nRows) {
  extern __shared__ char smem[];
  half_t* A0 = (half_t*)(smem + OFF_A0);
  int base = blockIdx.x * BROWS;
  for (int idx = threadIdx.x; idx < BROWS * 32; idx += TPB) {  // KP=32 (pad 4)
    int r = idx >> 5, k = idx & 31;
    int row = base + r;
    float v = (row < nRows && k < 4) ? efeat[row * 4 + k] : 0.f;
    A0[idx] = (half_t)v;
  }
  mlp_core<32, 4>(P, 4, 64, smem);
  float* FO = (float*)(smem + OFF_A0);
  int r = threadIdx.x;
  int row = base + r;
  if (row < nRows) {
    const float* f = FO + r * 64;
    float mu, inv; ln_stats(f, mu, inv);
    for (int c = 0; c < 64; ++c)
      e[row * 64 + c] = (half_t)((f[c] - mu) * inv * P.g[c] + P.b[c]);
  }
}

__global__ __launch_bounds__(TPB) void edge_kernel(half_t* e, const half_t* h, const int* src,
                                                   const int* dst, MlpP P, float* peSum,
                                                   int nRows) {
  extern __shared__ char smem[];
  half_t* A0 = (half_t*)(smem + OFF_A0);
  int base = blockIdx.x * BROWS;
  {  // gather-concat [e | h[src] | h[dst]] : 1 thread per row, 24 x uint4 (b128) copies
    int r = threadIdx.x;
    int row = base + r;
    bool valid = row < nRows;
    int rr = valid ? row : 0;
    int s = src[rr], d = dst[rr];
    const uint4* pe4 = (const uint4*)(e + (size_t)rr * 64);
    const uint4* ps4 = (const uint4*)(h + (size_t)s * 64);
    const uint4* pd4 = (const uint4*)(h + (size_t)d * 64);
    uint4* a4 = (uint4*)(A0 + r * 192);
    if (valid) {
#pragma unroll
      for (int q = 0; q < 8; ++q) a4[q] = pe4[q];
#pragma unroll
      for (int q = 0; q < 8; ++q) a4[8 + q] = ps4[q];
#pragma unroll
      for (int q = 0; q < 8; ++q) a4[16 + q] = pd4[q];
    } else {
      uint4 z = {0u, 0u, 0u, 0u};
#pragma unroll
      for (int q = 0; q < 24; ++q) a4[q] = z;
    }
  }
  mlp_core<192, 4>(P, 192, 64, smem);
  float* FO = (float*)(smem + OFF_A0);
  int r = threadIdx.x;
  int row = base + r;
  if (row < nRows) {
    const float* f = FO + r * 64;
    float mu, inv; ln_stats(f, mu, inv);
    int d = dst[row];
    for (int c = 0; c < 64; ++c) {
      float y = (f[c] - mu) * inv * P.g[c] + P.b[c] + (float)e[row * 64 + c];
      e[row * 64 + c] = (half_t)y;            // residual, in-place (own rows only)
      atomicAdd(&peSum[d * 64 + c], y);       // fused segment_sum (L2-resident)
    }
  }
}

__global__ __launch_bounds__(TPB) void node_kernel(half_t* h, const float* peSum, MlpP P,
                                                   int nRows) {
  extern __shared__ char smem[];
  half_t* A0 = (half_t*)(smem + OFF_A0);
  int base = blockIdx.x * BROWS;
  {  // [h | pe_sum], KP=128 ; 1 thread per row
    int r = threadIdx.x;
    int row = base + r;
    bool valid = row < nRows;
    uint4* a4 = (uint4*)(A0 + r * 128);
    if (valid) {
      const uint4* ph4 = (const uint4*)(h + (size_t)row * 64);
#pragma unroll
      for (int q = 0; q < 8; ++q) a4[q] = ph4[q];
      const float* pp = peSum + (size_t)row * 64;
      for (int k = 0; k < 64; ++k) A0[r * 128 + 64 + k] = (half_t)pp[k];
    } else {
      uint4 z = {0u, 0u, 0u, 0u};
#pragma unroll
      for (int q = 0; q < 16; ++q) a4[q] = z;
    }
  }
  mlp_core<128, 4>(P, 128, 64, smem);
  float* FO = (float*)(smem + OFF_A0);
  int r = threadIdx.x;
  int row = base + r;
  if (row < nRows) {
    const float* f = FO + r * 64;
    float mu, inv; ln_stats(f, mu, inv);
    for (int c = 0; c < 64; ++c) {
      float y = (f[c] - mu) * inv * P.g[c] + P.b[c] + (float)h[row * 64 + c];
      h[row * 64 + c] = (half_t)y;  // residual, in-place
    }
  }
}

__global__ __launch_bounds__(TPB) void out_kernel(const half_t* h, MlpP P, float* out,
                                                  int nRows) {
  extern __shared__ char smem[];
  half_t* A0 = (half_t*)(smem + OFF_A0);
  int base = blockIdx.x * BROWS;
  {  // KP=64 ; 1 thread per row
    int r = threadIdx.x;
    int row = base + r;
    uint4* a4 = (uint4*)(A0 + r * 64);
    if (row < nRows) {
      const uint4* ph4 = (const uint4*)(h + (size_t)row * 64);
#pragma unroll
      for (int q = 0; q < 8; ++q) a4[q] = ph4[q];
    } else {
      uint4 z = {0u, 0u, 0u, 0u};
#pragma unroll
      for (int q = 0; q < 8; ++q) a4[q] = z;
    }
  }
  mlp_core<64, 1>(P, 64, 2, smem);  // N padded to 16, only cols 0..1 valid
  float* FO = (float*)(smem + OFF_A0);
  int r = threadIdx.x;
  int row = base + r;
  if (row < nRows) {
    out[row * 2 + 0] = FO[r * 16 + 0];
    out[row * 2 + 1] = FO[r * 16 + 1];
  }
}

__global__ void zero_kernel(float* p, int n) {
  int i = blockIdx.x * blockDim.x + threadIdx.x;
  if (i < n) p[i] = 0.f;
}

// ---------------- host ----------------
// d_in order (jax pytree, dicts flatten by sorted key):
// 0 nfeatures, 1 next_flowrate, 2 inlet_mask, 3 efeatures, 4 src, 5 dst,
// params: enc_e(6..15), enc_n(16..25), out(26..33),
//         proc_e[0..4] at 34+10i, proc_n[0..4] at 84+10i.
// per normalized MLP: Wh0,Wh1,Wi,Wo,b,bh0,bh1,bi,bo,g ; out MLP: Wh0,Wh1,Wi,Wo,bh0,bh1,bi,bo.
extern "C" void kernel_launch(void* const* d_in, const int* in_sizes, int n_in,
                              void* d_out, int out_size, void* d_ws, size_t ws_size,
                              hipStream_t stream) {
  (void)in_sizes; (void)n_in; (void)out_size; (void)ws_size;
  const float* nfeat = (const float*)d_in[0];
  const float* flow  = (const float*)d_in[1];
  const int*   inlet = (const int*)d_in[2];
  const float* efeat = (const float*)d_in[3];
  const int*   src   = (const int*)d_in[4];
  const int*   dst   = (const int*)d_in[5];

  auto F = [&](int i) { return (const float*)d_in[i]; };
  auto mk = [&](int b, bool ln) {
    MlpP p;
    p.Wh0 = F(b + 0); p.Wh1 = F(b + 1); p.Wi = F(b + 2); p.Wo = F(b + 3);
    if (ln) { p.b = F(b + 4); p.bh0 = F(b + 5); p.bh1 = F(b + 6);
              p.bi = F(b + 7); p.bo = F(b + 8); p.g = F(b + 9); }
    else    { p.bh0 = F(b + 4); p.bh1 = F(b + 5); p.bi = F(b + 6);
              p.bo = F(b + 7); p.g = nullptr; p.b = nullptr; }
    return p;
  };
  MlpP enc_e = mk(6, true), enc_n = mk(16, true), outp = mk(26, false);
  MlpP pe_[5], pn_[5];
  for (int i = 0; i < 5; ++i) { pe_[i] = mk(34 + 10 * i, true); pn_[i] = mk(84 + 10 * i, true); }

  char* ws = (char*)d_ws;
  half_t* h     = (half_t*)ws;                                             // 6.4 MB
  half_t* e     = (half_t*)(ws + (size_t)NNODES * 64 * sizeof(half_t));    // 102.4 MB
  float*  peSum = (float*)(ws + (size_t)NNODES * 64 * sizeof(half_t)
                              + (size_t)NEDGES * 64 * sizeof(half_t));     // 12.8 MB

  const int gN = (NNODES + BROWS - 1) / BROWS;  // 196
  const int gE = (NEDGES + BROWS - 1) / BROWS;  // 3125

  enc_node_kernel<<<gN, TPB, SMEM_BYTES, stream>>>(nfeat, flow, inlet, enc_n, h, NNODES);
  enc_edge_kernel<<<gE, TPB, SMEM_BYTES, stream>>>(efeat, enc_e, e, NEDGES);

  for (int it = 0; it < 5; ++it) {
    zero_kernel<<<(NNODES * 64 + TPB - 1) / TPB, TPB, 0, stream>>>(peSum, NNODES * 64);
    edge_kernel<<<gE, TPB, SMEM_BYTES, stream>>>(e, h, src, dst, pe_[it], peSum, NEDGES);
    node_kernel<<<gN, TPB, SMEM_BYTES, stream>>>(h, peSum, pn_[it], NNODES);
  }
  out_kernel<<<gN, TPB, SMEM_BYTES, stream>>>(h, outp, (float*)d_out, NNODES);
}
